// GATLayer_76278619177408
// MI455X (gfx1250) — matrix operand order repaired and checked by reading, not compiled, not run on hardware
//
#include <hip/hip_runtime.h>

typedef __attribute__((ext_vector_type(2)))  float     v2f;
typedef __attribute__((ext_vector_type(8)))  float     v8f;
typedef __attribute__((ext_vector_type(2)))  _Float16  v2h;
typedef __attribute__((ext_vector_type(4)))  _Float16  v4h;
typedef __attribute__((ext_vector_type(8)))  _Float16  v8h;
typedef __attribute__((ext_vector_type(16))) _Float16  v16h;
typedef __attribute__((__vector_size__(4 * sizeof(int)))) int v4i;

#define GLOBAL_AS __attribute__((address_space(1)))
#define LDS_AS    __attribute__((address_space(3)))

#define BM   64      // block tile rows
#define KC   32      // K chunk (both kernels)
#define ASTR 36      // f32 A-tile LDS stride (kernel 1)
#define BSTR 264     // f32 B-tile LDS stride (kernel 1)
#define ASTR3 40     // f16 adj-tile LDS stride in halves (kernel 3)
#define BSTR3 40     // f16 h-tile per-column LDS stride in halves (kernel 3)

#if defined(__has_builtin)
#if __has_builtin(__builtin_amdgcn_global_load_async_to_lds_b128)
#define HAVE_ASYNC_LDS 1
#endif
#endif
#ifndef HAVE_ASYNC_LDS
#define HAVE_ASYNC_LDS 0
#endif

#if defined(__has_builtin)
#if __has_builtin(__builtin_amdgcn_s_wait_asynccnt)
#define WAIT_ASYNC(n) __builtin_amdgcn_s_wait_asynccnt(n)
#endif
#endif
#ifndef WAIT_ASYNC
#define WAIT_ASYNC(n) asm volatile("s_wait_asynccnt %0" ::"i"(n) : "memory")
#endif

// ======================= f32 staging helpers (kernel 1) =======================
__device__ __forceinline__ void load_chunk_A(const float* __restrict__ A, int ldA,
                                             int blockRow, int kc, int tid, float4 aReg[2]) {
#pragma unroll
  for (int i = 0; i < 2; ++i) {
    int idx = tid + i * 256, row = idx >> 3, col = (idx & 7) << 2;
    aReg[i] = *reinterpret_cast<const float4*>(A + (size_t)(blockRow + row) * ldA + kc + col);
  }
}
__device__ __forceinline__ void load_chunk_B(const float* __restrict__ B,
                                             int kc, int tid, float4 bReg[8]) {
#pragma unroll
  for (int i = 0; i < 8; ++i) {
    int idx = tid + i * 256, row = idx >> 6, col = (idx & 63) << 2;
    bReg[i] = *reinterpret_cast<const float4*>(B + (size_t)(kc + row) * 256 + col);
  }
}
__device__ __forceinline__ void store_chunk(const float4 aReg[2], const float4 bReg[8],
                                            float* As, float* Bs, int tid) {
#pragma unroll
  for (int i = 0; i < 2; ++i) {
    int idx = tid + i * 256, row = idx >> 3, col = (idx & 7) << 2;
    *reinterpret_cast<float4*>(&As[row * ASTR + col]) = aReg[i];
  }
#pragma unroll
  for (int i = 0; i < 8; ++i) {
    int idx = tid + i * 256, row = idx >> 6, col = (idx & 63) << 2;
    *reinterpret_cast<float4*>(&Bs[row * BSTR + col]) = bReg[i];
  }
}

// ---- f32 WMMA chunk: wave computes 16 rows x 128 cols with 16x16x4 f32 ----
__device__ __forceinline__ void mma_chunk_f32(const float* As, const float* Bs,
                                              int mg, int ng, int lane, v8f acc[8]) {
  const int half = lane >> 4, mrow = lane & 15;
  v2f af[8];
  const float* Ab = As + (mg * 16 + mrow) * ASTR + 2 * half;
#pragma unroll
  for (int ks = 0; ks < 8; ++ks) af[ks] = *reinterpret_cast<const v2f*>(Ab + ks * 4);
#pragma unroll
  for (int st = 0; st < 8; ++st) {
    const int col = ng * 128 + st * 16 + mrow;
    const float* Bb = Bs + col + 2 * half * BSTR;
#pragma unroll
    for (int ks = 0; ks < 8; ++ks) {
      v2f bf;
      bf.x = Bb[(ks * 4) * BSTR];
      bf.y = Bb[(ks * 4 + 1) * BSTR];
      acc[st] = __builtin_amdgcn_wmma_f32_16x16x4_f32(false, af[ks], false, bf,
                                                      (short)0, acc[st], false, false);
    }
  }
}

// ================= Kernel 1: h = X @ W  (M=16384, N=256, K=256), exact f32 =================
#if HAVE_ASYNC_LDS
__device__ __forceinline__ void async_stage(const float* __restrict__ X,
                                            const float* __restrict__ W,
                                            int blockRow, int kc, int tid,
                                            float* As, float* Bs) {
#pragma unroll
  for (int i = 0; i < 2; ++i) {
    int idx = tid + i * 256, row = idx >> 3, col = (idx & 7) << 2;
    __builtin_amdgcn_global_load_async_to_lds_b128(
        (GLOBAL_AS v4i*)(X + (size_t)(blockRow + row) * 256 + kc + col),
        (LDS_AS v4i*)&As[row * ASTR + col], 0, 0);
  }
#pragma unroll
  for (int i = 0; i < 8; ++i) {
    int idx = tid + i * 256, row = idx >> 6, col = (idx & 63) << 2;
    __builtin_amdgcn_global_load_async_to_lds_b128(
        (GLOBAL_AS v4i*)(W + (size_t)(kc + row) * 256 + col),
        (LDS_AS v4i*)&Bs[row * BSTR + col], 0, 0);
  }
}
#endif

__global__ __launch_bounds__(256) void gemm_xw_kernel(const float* __restrict__ X,
                                                      const float* __restrict__ W,
                                                      float* __restrict__ H) {
  const int tid = threadIdx.x, lane = tid & 31, wv = tid >> 5;
  const int mg = wv >> 1, ng = wv & 1;
  const int blockRow = blockIdx.x * BM;
  v8f acc[8];
#pragma unroll
  for (int i = 0; i < 8; ++i)
#pragma unroll
    for (int j = 0; j < 8; ++j) acc[i][j] = 0.0f;
  const int NCH = 256 / KC;

#if HAVE_ASYNC_LDS
  __shared__ __align__(16) float As[2][BM * ASTR];
  __shared__ __align__(16) float Bs[2][KC * BSTR];
  async_stage(X, W, blockRow, 0, tid, As[0], Bs[0]);
  for (int c = 0; c < NCH; ++c) {
    if (c > 0) __syncthreads();                       // buffer (c+1)&1 free to overwrite
    if (c + 1 < NCH) {
      async_stage(X, W, blockRow, (c + 1) * KC, tid, As[(c + 1) & 1], Bs[(c + 1) & 1]);
      WAIT_ASYNC(10);                                 // 10 in flight for c+1; chunk c done
    } else {
      WAIT_ASYNC(0);
    }
    __syncthreads();                                  // all waves' chunk-c copies visible
    mma_chunk_f32(As[c & 1], Bs[c & 1], mg, ng, lane, acc);
  }
#else
  __shared__ __align__(16) float As[BM * ASTR];
  __shared__ __align__(16) float Bs[KC * BSTR];
  float4 aReg[2], bReg[8];
  load_chunk_A(X, 256, blockRow, 0, tid, aReg);
  load_chunk_B(W, 0, tid, bReg);
  for (int c = 0; c < NCH; ++c) {
    __syncthreads();
    store_chunk(aReg, bReg, As, Bs, tid);
    __syncthreads();
    if (c + 1 < NCH) {
      load_chunk_A(X, 256, blockRow, (c + 1) * KC, tid, aReg);
      load_chunk_B(W, (c + 1) * KC, tid, bReg);
    }
    mma_chunk_f32(As, Bs, mg, ng, lane, acc);
  }
#endif

  const int half = lane >> 4, mrow = lane & 15;
#pragma unroll
  for (int st = 0; st < 8; ++st) {
    const int col = ng * 128 + st * 16 + mrow;
#pragma unroll
    for (int v = 0; v < 8; ++v) {
      const int row = blockRow + mg * 16 + v + 8 * half;
      H[(size_t)row * 256 + col] = acc[st][v];
    }
  }
}

// ================= Kernel 2: column sums of h (deg==0 fallback) =================
__global__ __launch_bounds__(256) void colsum_kernel(const float* __restrict__ H,
                                                     float* __restrict__ CS) {
  const int b = blockIdx.x, rg = blockIdx.y, f = threadIdx.x;
  const float* p = H + (size_t)b * 2048 * 256 + (size_t)rg * 256 * 256 + f;
  float s = 0.0f;
#pragma unroll 4
  for (int j = 0; j < 256; ++j) s += p[(size_t)j * 256];
  atomicAdd(&CS[b * 256 + f], s);
}

// ====== Kernel 3: out = elu( (adj @ h) / deg ), f16-split WMMA (exact-split h) ======
// adj is 0/1 -> exact in f16.  h = hi + lo (two f16 planes) -> adj@h = adj@hi + adj@lo,
// both accumulated in f32 by v_wmma_f32_16x16x32_f16 into the same accumulator.
__global__ __launch_bounds__(256) void gemm_att_kernel(const float* __restrict__ ADJ,
                                                       const float* __restrict__ H,
                                                       const float* __restrict__ CS,
                                                       float* __restrict__ OUT) {
  // A tile: rows stored in natural K order [0..31]
  __shared__ __align__(16) _Float16 As3[BM * ASTR3];
  // B tiles: column-major, (k,k+1) half-pairs packed -> frag = 2 contiguous b128 loads
  __shared__ __align__(16) _Float16 Bhi[256 * BSTR3];
  __shared__ __align__(16) _Float16 Blo[256 * BSTR3];

  const int tid = threadIdx.x, lane = tid & 31, wv = tid >> 5;
  const int mg = wv >> 1, ng = wv & 1;
  const int b = blockIdx.y, blockRow = blockIdx.x * BM;
  const float* A  = ADJ + (size_t)b * 2048 * 2048;
  const float* Hb = H + (size_t)b * 2048 * 256;

  v8f acc[8];
#pragma unroll
  for (int i = 0; i < 8; ++i)
#pragma unroll
    for (int j = 0; j < 8; ++j) acc[i][j] = 0.0f;
  float degacc = 0.0f;

  const int half = lane >> 4, mrow = lane & 15;

  float4 aR[2], bR0[4], bR1[4];
  load_chunk_A(A, 2048, blockRow, 0, tid, aR);
#pragma unroll
  for (int i = 0; i < 4; ++i) {                       // h rows 2*kp2 and 2*kp2+1
    int u = tid + i * 256, kp2 = u >> 6, c4 = (u & 63) << 2;
    bR0[i] = *reinterpret_cast<const float4*>(Hb + (size_t)(2 * kp2) * 256 + c4);
    bR1[i] = *reinterpret_cast<const float4*>(Hb + (size_t)(2 * kp2 + 1) * 256 + c4);
  }

  const int NCH = 2048 / KC;  // 64
  for (int c = 0; c < NCH; ++c) {
    __syncthreads();
    // ---- stage A (adj f32 -> f16, natural K order) ----
#pragma unroll
    for (int i = 0; i < 2; ++i) {
      int idx = tid + i * 256, row = idx >> 3, k4 = (idx & 7) << 2;
      int kp = k4;
      v4h ah = {(_Float16)aR[i].x, (_Float16)aR[i].y, (_Float16)aR[i].z, (_Float16)aR[i].w};
      *reinterpret_cast<v4h*>(&As3[row * ASTR3 + kp]) = ah;
    }
    // ---- stage B (h f32 -> hi/lo f16, column-major, k-pair packed) ----
#pragma unroll
    for (int i = 0; i < 4; ++i) {
      int u = tid + i * 256, kp2 = u >> 6, c4 = (u & 63) << 2;
      const float* f0 = reinterpret_cast<const float*>(&bR0[i]);
      const float* f1 = reinterpret_cast<const float*>(&bR1[i]);
#pragma unroll
      for (int j = 0; j < 4; ++j) {
        float x0 = f0[j], x1 = f1[j];
        _Float16 h0 = (_Float16)x0, h1 = (_Float16)x1;
        v2h hp = {h0, h1};
        v2h lp = {(_Float16)(x0 - (float)h0), (_Float16)(x1 - (float)h1)};
        *reinterpret_cast<v2h*>(&Bhi[(c4 + j) * BSTR3 + 2 * kp2]) = hp;
        *reinterpret_cast<v2h*>(&Blo[(c4 + j) * BSTR3 + 2 * kp2]) = lp;
      }
    }
    __syncthreads();
    // ---- prefetch next chunk into registers ----
    if (c + 1 < NCH) {
      int kc = (c + 1) * KC;
      load_chunk_A(A, 2048, blockRow, kc, tid, aR);
#pragma unroll
      for (int i = 0; i < 4; ++i) {
        int u = tid + i * 256, kp2 = u >> 6, c4 = (u & 63) << 2;
        bR0[i] = *reinterpret_cast<const float4*>(Hb + (size_t)(kc + 2 * kp2) * 256 + c4);
        bR1[i] = *reinterpret_cast<const float4*>(Hb + (size_t)(kc + 2 * kp2 + 1) * 256 + c4);
      }
    }
    // ---- compute: 8 subtiles x (hi + lo) WMMA ----
    const _Float16* Ab = &As3[(mg * 16 + mrow) * ASTR3 + 16 * half];
    v8h a0 = *reinterpret_cast<const v8h*>(Ab);
    v8h a1 = *reinterpret_cast<const v8h*>(Ab + 8);
    v16h aF = __builtin_shufflevector(a0, a1, 0, 1, 2, 3, 4, 5, 6, 7,
                                      8, 9, 10, 11, 12, 13, 14, 15);
#pragma unroll
    for (int i = 0; i < 8; ++i) degacc += (float)a0[i] + (float)a1[i];  // adj row-sum
#pragma unroll
    for (int st = 0; st < 8; ++st) {
      const int col = ng * 128 + st * 16 + mrow;
      const _Float16* Bp = &Bhi[col * BSTR3 + 16 * half];
      v8h b0 = *reinterpret_cast<const v8h*>(Bp);
      v8h b1 = *reinterpret_cast<const v8h*>(Bp + 8);
      v16h bH = __builtin_shufflevector(b0, b1, 0, 1, 2, 3, 4, 5, 6, 7,
                                        8, 9, 10, 11, 12, 13, 14, 15);
      const _Float16* Lp = &Blo[col * BSTR3 + 16 * half];
      v8h l0 = *reinterpret_cast<const v8h*>(Lp);
      v8h l1 = *reinterpret_cast<const v8h*>(Lp + 8);
      v16h bL = __builtin_shufflevector(l0, l1, 0, 1, 2, 3, 4, 5, 6, 7,
                                        8, 9, 10, 11, 12, 13, 14, 15);
      acc[st] = __builtin_amdgcn_wmma_f32_16x16x32_f16(false, aF, false, bH,
                                                       (short)0, acc[st], false, false);
      acc[st] = __builtin_amdgcn_wmma_f32_16x16x32_f16(false, aF, false, bL,
                                                       (short)0, acc[st], false, false);
    }
  }

  // full row degree: combine the two lane-halves
  const float degRow = degacc + __shfl_xor(degacc, 16);
#pragma unroll
  for (int st = 0; st < 8; ++st) {
    const int col = ng * 128 + st * 16 + mrow;
    const float csv = CS[b * 256 + col] * (1.0f / 2048.0f);   // uniform-softmax fallback
#pragma unroll
    for (int v = 0; v < 8; ++v) {
      const float d = __shfl(degRow, v + 8 * half);           // deg of row (v + 8*half)
      float x = (d > 0.0f) ? acc[st][v] / d : csv;
      x = (x > 0.0f) ? x : (expf(x) - 1.0f);                  // ELU
      const int row = blockRow + mg * 16 + v + 8 * half;
      OUT[(size_t)b * 2048 * 256 + (size_t)row * 256 + col] = x;
    }
  }
}

extern "C" void kernel_launch(void* const* d_in, const int* in_sizes, int n_in,
                              void* d_out, int out_size, void* d_ws, size_t ws_size,
                              hipStream_t stream) {
  (void)in_sizes; (void)n_in; (void)out_size; (void)ws_size;
  const float* X   = (const float*)d_in[0];   // (8,2048,256)
  const float* ADJ = (const float*)d_in[1];   // (8,2048,2048)
  const float* W   = (const float*)d_in[2];   // (256,256)
  // d_in[3] = a : cancels exactly (softmax of a per-row constant is uniform over neighbors)
  float* OUT = (float*)d_out;                 // (8,2048,256)

  float* H  = (float*)d_ws;                   // 16 MB
  float* CS = H + (size_t)8 * 2048 * 256;     // 8 KB

  (void)hipMemsetAsync(CS, 0, (size_t)8 * 256 * sizeof(float), stream);
  gemm_xw_kernel<<<dim3(16384 / BM), 256, 0, stream>>>(X, W, H);
  colsum_kernel<<<dim3(8, 8), 256, 0, stream>>>(H, CS);
  gemm_att_kernel<<<dim3(2048 / BM, 8), 256, 0, stream>>>(ADJ, H, CS, OUT);
}
